// Detect_Head_66400194396314
// MI455X (gfx1250) — matrix operand order repaired and checked by reading, not compiled, not run on hardware
//
#include <hip/hip_runtime.h>
#include <hip/hip_bf16.h>
#include <math.h>

typedef __bf16 bf16_t;
typedef __attribute__((ext_vector_type(16))) bf16_t       v16bf;
typedef __attribute__((ext_vector_type(8)))  float        v8f;
typedef __attribute__((ext_vector_type(4)))  unsigned int v4u;
typedef __attribute__((ext_vector_type(4)))  unsigned int s4u;
typedef __attribute__((ext_vector_type(8)))  unsigned int s8u;

union Frag { v16bf bf; v4u u[2]; };

#define N_INST 1024
#define C_IN   512
#define C_MID  1024
#define SP     49
#define K1     (C_IN * 9)    /* 4608 */
#define K2     (C_MID * 9)   /* 9216 */
#define KD     C_IN          /* 512  */
#define NCLS   9
#define SCORE_TH 0.05f
#define NMS_TH   0.5f
#define MIN_SIZE 0.01f
#define DETS     100
#define CAND     4096
#define BBOX_CLIP 4.135166556742356f

// d_out float offsets (concatenated return tuple)
#define EMB_OFF  0UL
#define CLS_OFF  51380224UL
#define REG_OFF  51389440UL
#define DBOX_OFF 51426304UL
#define DSC_OFF  51427104UL
#define DLB_OFF  51427304UL
#define DVL_OFF  51427504UL

__device__ __forceinline__ unsigned short f2bf(float f) {
    unsigned int u = __float_as_uint(f);
    u += 0x7FFFu + ((u >> 16) & 1u);
    return (unsigned short)(u >> 16);
}

__global__ void cvt_bf16(const float* __restrict__ src,
                         unsigned short* __restrict__ dst, int n) {
    int i = blockIdx.x * 256 + threadIdx.x;
    if (i < n) dst[i] = f2bf(src[i]);
}

// repack [O,C,3,3] f32 -> [O, rs*C + ic] bf16 (rs-major K so the hot loop has
// uniform (r,s) per 32-wide K step and pure stride gathers)
__global__ void repack_w_bf16(const float* __restrict__ src,
                              unsigned short* __restrict__ dst, int C, int total) {
    int i = blockIdx.x * 256 + threadIdx.x;
    if (i >= total) return;
    int K  = C * 9;
    int oc = i / K;
    int rem = i - oc * K;
    int ic = rem / 9;
    int rs = rem - ic * 9;
    dst[(size_t)oc * K + rs * C + ic] = f2bf(src[i]);
}

// ---- Tensor Data Mover: async 2D tile (rows x 32 bf16) -> LDS with 16B row pad ----
// D# group0/group1 built per CDNA5 ISA.
// pad_enable, pad_interval=16 DWORDs (code 3), pad_amount=4 DWORDs (code 3)
// => LDS row stride = 64B data + 16B pad = 80B = 40 ushorts.
__device__ __forceinline__ void tdm_load_a(const unsigned short* gptr,
                                           unsigned rowlen_elems,
                                           unsigned nrows,
                                           unsigned lds_byte_addr) {
    unsigned long ga = (unsigned long)gptr;
    s4u g0;
    g0[0] = 1u;                                   // count=1 (valid), user mode
    g0[1] = lds_byte_addr;                        // lds_addr
    g0[2] = (unsigned)ga;                         // global_addr[31:0]
    g0[3] = (unsigned)(ga >> 32) | (2u << 30);    // global_addr[56:32] | type=2
    s8u g1;
    g1[0] = (1u << 16) | (1u << 20) | (3u << 22) | (3u << 25); // data_size=2B, pad en/int/amt
    g1[1] = (rowlen_elems & 0xFFFFu) << 16;       // tensor_dim0[15:0]
    g1[2] = (rowlen_elems >> 16) | (1024u << 16); // tensor_dim0[31:16] | tensor_dim1[15:0]
    g1[3] = (32u << 16);                          // tensor_dim1[31:16]=0 | tile_dim0=32
    g1[4] = nrows;                                // tile_dim1 | tile_dim2=0
    g1[5] = rowlen_elems;                         // tensor_dim0_stride[31:0]
    g1[6] = 0u;                                   // stride hi | tensor_dim1_stride lo
    g1[7] = 0u;
    asm volatile("tensor_load_to_lds %0, %1" :: "s"(g0), "s"(g1) : "memory");
}

// ---------------- conv1: h = relu(conv3x3(x, w1)) implicit GEMM, M=256/block ----------------
__global__ __launch_bounds__(256) void conv1_wmma(
    const unsigned short* __restrict__ xb,   // [1024,512,49] bf16
    const unsigned short* __restrict__ w1b,  // [1024, rs*512+ic] bf16
    unsigned short* __restrict__ hb)         // [1024,1024,49] bf16
{
    __shared__ unsigned short lds_a[2][256 * 40];
    __shared__ unsigned short lds_b[64 * 40];
    const int img   = blockIdx.y;
    const int mbase = blockIdx.x * 256;
    const int tid   = threadIdx.x;
    const int lane  = tid & 31, wave = tid >> 5;

    v8f acc[2][4];
    for (int h = 0; h < 2; ++h)
        for (int i = 0; i < 4; ++i)
            for (int r = 0; r < 8; ++r) acc[h][i][r] = 0.0f;

    const int bp = tid >> 2;            // 0..63 spatial position
    const int bk = (tid & 3) * 8;       // ic sub-offset
    const int oy = bp / 7, ox = bp - (bp / 7) * 7;
    const unsigned short* xi = xb + (size_t)img * C_IN * SP;

    if (wave == 0)
        tdm_load_a(w1b + (size_t)mbase * K1, K1, 256, (unsigned)(size_t)&lds_a[0][0]);

    for (int kk = 0; kk < K1; kk += 32) {
        const int buf = (kk >> 5) & 1;
        // ---- B tile gather (im2col, rs uniform per step) ----
        {
            const int rs = kk >> 9;               // k = rs*512 + ic
            const int r = rs / 3, s = rs - r * 3;
            const int iy = oy + r - 1, ix = ox + s - 1;
            const bool ok = (bp < SP) && ((unsigned)iy < 7u) && ((unsigned)ix < 7u);
            const unsigned short* src = xi + (size_t)((kk & 511) + bk) * SP + iy * 7 + ix;
            unsigned int vals[8] = {0, 0, 0, 0, 0, 0, 0, 0};
            if (ok) {
                #pragma unroll
                for (int i = 0; i < 8; ++i) vals[i] = src[i * SP];
            }
            v4u pk;
            pk[0] = vals[0] | (vals[1] << 16);
            pk[1] = vals[2] | (vals[3] << 16);
            pk[2] = vals[4] | (vals[5] << 16);
            pk[3] = vals[6] | (vals[7] << 16);
            *(v4u*)&lds_b[bp * 40 + bk] = pk;
        }
        if (wave == 0) __builtin_amdgcn_s_wait_tensorcnt(0);
        __syncthreads();
        if (wave == 0 && kk + 32 < K1)
            tdm_load_a(w1b + (size_t)mbase * K1 + kk + 32, K1, 256,
                       (unsigned)(size_t)&lds_a[buf ^ 1][0]);
        // ---- compute: 2 M-frags x 4 N-frags = 8 WMMA ----
        Frag a0, a1;
        {
            const int row = wave * 16 + (lane & 15);
            const int kb  = (lane >> 4) * 8;
            a0.u[0] = *(const v4u*)&lds_a[buf][row * 40 + kb];
            a0.u[1] = *(const v4u*)&lds_a[buf][row * 40 + kb + 16];
            a1.u[0] = *(const v4u*)&lds_a[buf][(row + 128) * 40 + kb];
            a1.u[1] = *(const v4u*)&lds_a[buf][(row + 128) * 40 + kb + 16];
        }
        #pragma unroll
        for (int nt = 0; nt < 4; ++nt) {
            Frag Bf;
            const int col = nt * 16 + (lane & 15);
            const int kb2 = (lane >> 4) * 8;
            Bf.u[0] = *(const v4u*)&lds_b[col * 40 + kb2];
            Bf.u[1] = *(const v4u*)&lds_b[col * 40 + kb2 + 16];
            acc[0][nt] = __builtin_amdgcn_wmma_f32_16x16x32_bf16(
                false, a0.bf, false, Bf.bf, (short)0, acc[0][nt], false, false);
            acc[1][nt] = __builtin_amdgcn_wmma_f32_16x16x32_bf16(
                false, a1.bf, false, Bf.bf, (short)0, acc[1][nt], false, false);
        }
        __syncthreads();
    }
    const int p0 = lane & 15;
    const int rsh = (lane >> 4) * 8;
    #pragma unroll
    for (int h = 0; h < 2; ++h)
        #pragma unroll
        for (int nt = 0; nt < 4; ++nt) {
            int p = nt * 16 + p0;
            if (p >= SP) continue;
            #pragma unroll
            for (int r = 0; r < 8; ++r) {
                int oc = mbase + h * 128 + wave * 16 + r + rsh;
                hb[((size_t)img * C_MID + oc) * SP + p] = f2bf(fmaxf(acc[h][nt][r], 0.0f));
            }
        }
}

// -------- conv2: embedded = relu(conv3x3(h,w2) + conv1x1(x,wd)), fused K loop --------
__global__ __launch_bounds__(256) void conv2_wmma(
    const unsigned short* __restrict__ xb,
    const unsigned short* __restrict__ hb,
    const unsigned short* __restrict__ w2b,  // [1024, rs*1024+ic]
    const unsigned short* __restrict__ wdb,  // [1024, 512]
    float* __restrict__ emb)                 // [1024,1024,49] f32 (d_out)
{
    __shared__ unsigned short lds_a[2][256 * 40];
    __shared__ unsigned short lds_b[64 * 40];
    const int img   = blockIdx.y;
    const int mbase = blockIdx.x * 256;
    const int tid   = threadIdx.x;
    const int lane  = tid & 31, wave = tid >> 5;

    v8f acc[2][4];
    for (int h = 0; h < 2; ++h)
        for (int i = 0; i < 4; ++i)
            for (int r = 0; r < 8; ++r) acc[h][i][r] = 0.0f;

    const int bp = tid >> 2;
    const int bk = (tid & 3) * 8;
    const int oy = bp / 7, ox = bp - (bp / 7) * 7;
    const unsigned short* xi = xb + (size_t)img * C_IN * SP;
    const unsigned short* hi = hb + (size_t)img * C_MID * SP;

    if (wave == 0)
        tdm_load_a(w2b + (size_t)mbase * K2, K2, 256, (unsigned)(size_t)&lds_a[0][0]);

    const int KT = K2 + KD;
    for (int kk = 0; kk < KT; kk += 32) {
        const int buf = (kk >> 5) & 1;
        // ---- B tile ----
        {
            unsigned int vals[8] = {0, 0, 0, 0, 0, 0, 0, 0};
            if (kk < K2) {                        // im2col(h), k = rs*1024 + ic
                const int rs = kk >> 10;
                const int r = rs / 3, s = rs - r * 3;
                const int iy = oy + r - 1, ix = ox + s - 1;
                const bool ok = (bp < SP) && ((unsigned)iy < 7u) && ((unsigned)ix < 7u);
                const unsigned short* src = hi + (size_t)((kk & 1023) + bk) * SP + iy * 7 + ix;
                if (ok) {
                    #pragma unroll
                    for (int i = 0; i < 8; ++i) vals[i] = src[i * SP];
                }
            } else {                              // 1x1 shortcut: B[k][p] = x[k][p]
                if (bp < SP) {
                    const unsigned short* src = xi + (size_t)(kk - K2 + bk) * SP + bp;
                    #pragma unroll
                    for (int i = 0; i < 8; ++i) vals[i] = src[i * SP];
                }
            }
            v4u pk;
            pk[0] = vals[0] | (vals[1] << 16);
            pk[1] = vals[2] | (vals[3] << 16);
            pk[2] = vals[4] | (vals[5] << 16);
            pk[3] = vals[6] | (vals[7] << 16);
            *(v4u*)&lds_b[bp * 40 + bk] = pk;
        }
        if (wave == 0) __builtin_amdgcn_s_wait_tensorcnt(0);
        __syncthreads();
        if (wave == 0 && kk + 32 < KT) {
            const int nk = kk + 32;
            if (nk < K2)
                tdm_load_a(w2b + (size_t)mbase * K2 + nk, K2, 256,
                           (unsigned)(size_t)&lds_a[buf ^ 1][0]);
            else
                tdm_load_a(wdb + (size_t)mbase * KD + (nk - K2), KD, 256,
                           (unsigned)(size_t)&lds_a[buf ^ 1][0]);
        }
        // ---- compute ----
        Frag a0, a1;
        {
            const int row = wave * 16 + (lane & 15);
            const int kb  = (lane >> 4) * 8;
            a0.u[0] = *(const v4u*)&lds_a[buf][row * 40 + kb];
            a0.u[1] = *(const v4u*)&lds_a[buf][row * 40 + kb + 16];
            a1.u[0] = *(const v4u*)&lds_a[buf][(row + 128) * 40 + kb];
            a1.u[1] = *(const v4u*)&lds_a[buf][(row + 128) * 40 + kb + 16];
        }
        #pragma unroll
        for (int nt = 0; nt < 4; ++nt) {
            Frag Bf;
            const int col = nt * 16 + (lane & 15);
            const int kb2 = (lane >> 4) * 8;
            Bf.u[0] = *(const v4u*)&lds_b[col * 40 + kb2];
            Bf.u[1] = *(const v4u*)&lds_b[col * 40 + kb2 + 16];
            acc[0][nt] = __builtin_amdgcn_wmma_f32_16x16x32_bf16(
                false, a0.bf, false, Bf.bf, (short)0, acc[0][nt], false, false);
            acc[1][nt] = __builtin_amdgcn_wmma_f32_16x16x32_bf16(
                false, a1.bf, false, Bf.bf, (short)0, acc[1][nt], false, false);
        }
        __syncthreads();
    }
    const int p0 = lane & 15;
    const int rsh = (lane >> 4) * 8;
    #pragma unroll
    for (int h = 0; h < 2; ++h)
        #pragma unroll
        for (int nt = 0; nt < 4; ++nt) {
            int p = nt * 16 + p0;
            if (p >= SP) continue;
            #pragma unroll
            for (int r = 0; r < 8; ++r) {
                int oc = mbase + h * 128 + wave * 16 + r + rsh;
                emb[((size_t)img * C_MID + oc) * SP + p] = fmaxf(acc[h][nt][r], 0.0f);
            }
        }
}

// ---------------- pool + FC heads + box decode + softmax ----------------
__global__ __launch_bounds__(256) void head_kernel(
    const float* __restrict__ emb, const float* __restrict__ proposals,
    const float* __restrict__ cls_w, const float* __restrict__ cls_b,
    const float* __restrict__ reg_w, const float* __restrict__ reg_b,
    float* __restrict__ cls_out, float* __restrict__ reg_out,
    float* __restrict__ pboxes, float* __restrict__ pscores)
{
    __shared__ float sflat[C_MID];
    __shared__ float out45[48];
    const int n = blockIdx.x;
    const int tid = threadIdx.x;
    const float* e = emb + (size_t)n * C_MID * SP;
    for (int c = tid; c < C_MID; c += 256) {
        float s = 0.f;
        const float* ec = e + (size_t)c * SP;
        for (int p = 0; p < SP; ++p) s += ec[p];
        sflat[c] = s * (1.0f / 49.0f);
    }
    __syncthreads();
    const int lane = tid & 31, wave = tid >> 5;
    for (int o = wave; o < 45; o += 8) {
        const float* w = (o < NCLS) ? (cls_w + (size_t)o * C_MID)
                                    : (reg_w + (size_t)(o - NCLS) * C_MID);
        float sum = 0.f;
        for (int k = lane; k < C_MID; k += 32) sum += sflat[k] * w[k];
        for (int m = 16; m > 0; m >>= 1) sum += __shfl_xor(sum, m, 32);
        if (lane == 0) out45[o] = sum + ((o < NCLS) ? cls_b[o] : reg_b[o - NCLS]);
    }
    __syncthreads();
    if (tid < NCLS)  cls_out[(size_t)n * NCLS + tid] = out45[tid];
    if (tid < 36)    reg_out[(size_t)n * 36 + tid]   = out45[NCLS + tid];
    if (tid == 0) {
        const float* pr = proposals + (size_t)n * 4;
        float pw = pr[2] - pr[0], ph = pr[3] - pr[1];
        float pcx = pr[0] + 0.5f * pw, pcy = pr[1] + 0.5f * ph;
        float mx = out45[0];
        for (int i = 1; i < NCLS; ++i) mx = fmaxf(mx, out45[i]);
        float se = 0.f, ex[NCLS];
        for (int i = 0; i < NCLS; ++i) { ex[i] = __expf(out45[i] - mx); se += ex[i]; }
        float inv = 1.0f / se;
        int img = n >> 9, pidx = n & 511;
        for (int c = 1; c < NCLS; ++c) {
            float dx = out45[NCLS + c * 4 + 0] * 0.1f;
            float dy = out45[NCLS + c * 4 + 1] * 0.1f;
            float dw = fminf(out45[NCLS + c * 4 + 2] * 0.2f, BBOX_CLIP);
            float dh = fminf(out45[NCLS + c * 4 + 3] * 0.2f, BBOX_CLIP);
            float ncx = dx * pw + pcx, ncy = dy * ph + pcy;
            float nw = __expf(dw) * pw, nh = __expf(dh) * ph;
            float x1 = fminf(fmaxf(ncx - 0.5f * nw, 0.f), 640.f);
            float y1 = fminf(fmaxf(ncy - 0.5f * nh, 0.f), 384.f);
            float x2 = fminf(fmaxf(ncx + 0.5f * nw, 0.f), 640.f);
            float y2 = fminf(fmaxf(ncy + 0.5f * nh, 0.f), 384.f);
            size_t ci = ((size_t)img * 512 + pidx) * 8 + (c - 1);
            pboxes[ci * 4 + 0] = x1; pboxes[ci * 4 + 1] = y1;
            pboxes[ci * 4 + 2] = x2; pboxes[ci * 4 + 3] = y2;
            pscores[ci] = ex[c] * inv;
        }
    }
}

// ---------------- per-image class-aware NMS ----------------
__global__ __launch_bounds__(256) void nms_kernel(
    const float* __restrict__ pboxes, const float* __restrict__ pscores,
    float* __restrict__ dout)
{
    __shared__ float key[CAND];
    __shared__ int   sidx[CAND];
    __shared__ float tb[512 * 4];
    __shared__ float ts[512];
    __shared__ int   tl[512];
    __shared__ int   keep[512];
    const int img = blockIdx.x;
    const int tid = threadIdx.x;
    const float* bb = pboxes + (size_t)img * 512 * 8 * 4;
    const float* ss = pscores + (size_t)img * 512 * 8;
    for (int j = tid; j < CAND; j += 256) {
        float s = ss[j];
        float w = bb[j * 4 + 2] - bb[j * 4 + 0];
        float h = bb[j * 4 + 3] - bb[j * 4 + 1];
        key[j] = (s > SCORE_TH && w >= MIN_SIZE && h >= MIN_SIZE) ? s : -INFINITY;
        sidx[j] = j;
    }
    __syncthreads();
    for (int k = 2; k <= CAND; k <<= 1) {           // bitonic sort, descending
        for (int j = k >> 1; j > 0; j >>= 1) {
            for (int i = tid; i < CAND; i += 256) {
                int ixj = i ^ j;
                if (ixj > i) {
                    bool desc = ((i & k) == 0);
                    float a = key[i], b = key[ixj];
                    bool sw = desc ? (a < b) : (a > b);
                    if (sw) {
                        key[i] = b; key[ixj] = a;
                        int t0 = sidx[i]; sidx[i] = sidx[ixj]; sidx[ixj] = t0;
                    }
                }
            }
            __syncthreads();
        }
    }
    for (int i = tid; i < 512; i += 256) {          // top-512 gather
        int j = sidx[i];
        tb[i * 4 + 0] = bb[j * 4 + 0]; tb[i * 4 + 1] = bb[j * 4 + 1];
        tb[i * 4 + 2] = bb[j * 4 + 2]; tb[i * 4 + 3] = bb[j * 4 + 3];
        ts[i] = key[i];
        tl[i] = (j & 7) + 1;
        keep[i] = (key[i] > -1e30f) ? 1 : 0;
    }
    __syncthreads();
    for (int i = 0; i < 512; ++i) {                 // greedy suppression
        if (keep[i]) {
            float ax1 = tb[i * 4], ay1 = tb[i * 4 + 1];
            float ax2 = tb[i * 4 + 2], ay2 = tb[i * 4 + 3];
            float aarea = (ax2 - ax1) * (ay2 - ay1);
            int la = tl[i];
            for (int j = i + 1 + tid; j < 512; j += 256) {
                if (keep[j] && tl[j] == la) {
                    float ix1 = fmaxf(ax1, tb[j * 4]),     iy1 = fmaxf(ay1, tb[j * 4 + 1]);
                    float ix2 = fminf(ax2, tb[j * 4 + 2]), iy2 = fminf(ay2, tb[j * 4 + 3]);
                    float iw = fmaxf(ix2 - ix1, 0.f), ih = fmaxf(iy2 - iy1, 0.f);
                    float inter = iw * ih;
                    float barea = (tb[j * 4 + 2] - tb[j * 4]) * (tb[j * 4 + 3] - tb[j * 4 + 1]);
                    float iou = inter / fmaxf(aarea + barea - inter, 1e-8f);
                    if (iou > NMS_TH) keep[j] = 0;
                }
            }
        }
        __syncthreads();
    }
    if (tid == 0) {
        int cnt = 0;
        for (int i = 0; i < 512 && cnt < DETS; ++i) {
            if (keep[i]) {
                size_t o = (size_t)img * DETS + cnt;
                dout[DBOX_OFF + o * 4 + 0] = tb[i * 4 + 0];
                dout[DBOX_OFF + o * 4 + 1] = tb[i * 4 + 1];
                dout[DBOX_OFF + o * 4 + 2] = tb[i * 4 + 2];
                dout[DBOX_OFF + o * 4 + 3] = tb[i * 4 + 3];
                dout[DSC_OFF + o] = ts[i];
                dout[DLB_OFF + o] = (float)tl[i];
                dout[DVL_OFF + o] = 1.0f;
                ++cnt;
            }
        }
        for (; cnt < DETS; ++cnt) {
            size_t o = (size_t)img * DETS + cnt;
            dout[DBOX_OFF + o * 4 + 0] = 0.f; dout[DBOX_OFF + o * 4 + 1] = 0.f;
            dout[DBOX_OFF + o * 4 + 2] = 0.f; dout[DBOX_OFF + o * 4 + 3] = 0.f;
            dout[DSC_OFF + o] = 0.f; dout[DLB_OFF + o] = 0.f; dout[DVL_OFF + o] = 0.f;
        }
    }
}

extern "C" void kernel_launch(void* const* d_in, const int* in_sizes, int n_in,
                              void* d_out, int out_size, void* d_ws, size_t ws_size,
                              hipStream_t stream)
{
    const float* feats     = (const float*)d_in[0];
    const float* proposals = (const float*)d_in[1];
    const float* w1        = (const float*)d_in[2];
    const float* w2        = (const float*)d_in[3];
    const float* wd        = (const float*)d_in[4];
    const float* cls_w     = (const float*)d_in[5];
    const float* cls_b     = (const float*)d_in[6];
    const float* reg_w     = (const float*)d_in[7];
    const float* reg_b     = (const float*)d_in[8];
    float* dout = (float*)d_out;
    char*  ws   = (char*)d_ws;

    unsigned short* xb  = (unsigned short*)(ws + 0);
    unsigned short* w1b = (unsigned short*)(ws + 51380224UL);
    unsigned short* w2b = (unsigned short*)(ws + 60817408UL);
    unsigned short* wdb = (unsigned short*)(ws + 79691776UL);
    unsigned short* hb  = (unsigned short*)(ws + 80740352UL);
    float* pboxes  = (float*)(ws + 183500800UL);
    float* pscores = (float*)(ws + 183631872UL);

    cvt_bf16<<<(25690112 + 255) / 256, 256, 0, stream>>>(feats, xb, 25690112);
    repack_w_bf16<<<(4718592 + 255) / 256, 256, 0, stream>>>(w1, w1b, C_IN, 4718592);
    repack_w_bf16<<<(9437184 + 255) / 256, 256, 0, stream>>>(w2, w2b, C_MID, 9437184);
    cvt_bf16<<<(524288 + 255) / 256, 256, 0, stream>>>(wd, wdb, 524288);

    conv1_wmma<<<dim3(4, N_INST), 256, 0, stream>>>(xb, w1b, hb);
    conv2_wmma<<<dim3(4, N_INST), 256, 0, stream>>>(xb, hb, w2b, wdb, dout + EMB_OFF);

    head_kernel<<<N_INST, 256, 0, stream>>>(dout + EMB_OFF, proposals,
                                            cls_w, cls_b, reg_w, reg_b,
                                            dout + CLS_OFF, dout + REG_OFF,
                                            pboxes, pscores);
    nms_kernel<<<2, 256, 0, stream>>>(pboxes, pscores, dout);
}
